// TripletLossAttrWeightes_36610301231654
// MI455X (gfx1250) — compile-verified, matrix operand order checked by reading, not run on hardware
//
#include <hip/hip_runtime.h>
#include <hip/hip_bf16.h>
#include <stdint.h>

#define NN   4096
#define DD   2048
#define DW   512
#define QCOLS 1024           // columns per block (4-way column split)
#define MARGIN_F 0.3f

typedef __attribute__((ext_vector_type(16))) __bf16 v16bf;
typedef __attribute__((ext_vector_type(8)))  __bf16 v8bf;
typedef __attribute__((ext_vector_type(8)))  float  v8f;
typedef unsigned long long u64;

// Gather one 16-bit WMMA A/B fragment for this lane from a bf16 plane row.
// Layout (ISA 7.12.2, 16-bit A 16x32): elem i of v16bf holds
//   K = (i&7) + ((i>>3)<<4) + ((lane>=16)?8:0)
// -> elems 0..7 and 8..15 are two contiguous 16B runs (two b128 loads).
__device__ __forceinline__ v16bf load_frag_bf(const __bf16* __restrict__ row,
                                              int kbase /* k0 + (lane>=16?8:0) */) {
  v8bf a = *(const v8bf*)(row + kbase);
  v8bf b = *(const v8bf*)(row + kbase + 16);
  return __builtin_shufflevector(a, b, 0, 1, 2, 3, 4, 5, 6, 7,
                                 8, 9, 10, 11, 12, 13, 14, 15);
}

// Map fp32 -> monotone u32 (total order matches float order).
__device__ __forceinline__ unsigned ord_u32(float v) {
  unsigned u = __float_as_uint(v);
  return (u & 0x80000000u) ? ~u : (u | 0x80000000u);
}

__device__ __forceinline__ u64 shfl_xor_u64(u64 v, int m) {
  unsigned loh = (unsigned)v;
  unsigned hih = (unsigned)(v >> 32);
  loh = __shfl_xor(loh, m, 32);
  hih = __shfl_xor(hih, m, 32);
  return ((u64)hih << 32) | loh;
}

// ---------------------------------------------------------------------------
// Kernel 0: init packed-best arrays (harness poisons ws; must be deterministic).
// ---------------------------------------------------------------------------
__global__ __launch_bounds__(256)
void init_best_kernel(u64* __restrict__ pmax, u64* __restrict__ pmin) {
  const int i = blockIdx.x * 256 + threadIdx.x;
  pmax[i] = 0ull;
  pmin[i] = ~0ull;
}

// ---------------------------------------------------------------------------
// Kernel 1: row norms xx[i]=|inputs[i]|^2, ww[i]=|w[i]|^2, and one-shot
// bf16 hi/lo split of w into L2-resident planes (x = hi + lo, RNE).
// ---------------------------------------------------------------------------
__global__ __launch_bounds__(256)
void norms_split_kernel(const float* __restrict__ x, const float* __restrict__ w,
                        float* __restrict__ xx, float* __restrict__ ww,
                        __bf16* __restrict__ w_hi, __bf16* __restrict__ w_lo) {
  const int row = blockIdx.x;
  const int tid = threadIdx.x;
  const float* xr = x + (size_t)row * DD;
  const float* wr = w + (size_t)row * DW;
  float sx = 0.f, sw = 0.f;
  for (int k = tid; k < DD; k += 256) { float v = xr[k]; sx += v * v; }
  for (int k = tid; k < DW; k += 256) {
    float v = wr[k];
    sw += v * v;
    __bf16 h = (__bf16)v;
    w_hi[(size_t)row * DW + k] = h;
    w_lo[(size_t)row * DW + k] = (__bf16)(v - (float)h);
  }
#pragma unroll
  for (int m = 16; m >= 1; m >>= 1) {
    sx += __shfl_xor(sx, m, 32);
    sw += __shfl_xor(sw, m, 32);
  }
  __shared__ float rx[8], rw[8];
  if ((tid & 31) == 0) { rx[tid >> 5] = sx; rw[tid >> 5] = sw; }
  __syncthreads();
  if (tid == 0) {
    float a = 0.f, b = 0.f;
#pragma unroll
    for (int i = 0; i < 8; ++i) { a += rx[i]; b += rw[i]; }
    xx[row] = a;
    ww[row] = b;
  }
}

// ---------------------------------------------------------------------------
// Kernel 2: fused w-Gram (3x bf16-split WMMA, fp32 accumulate) + masked
// argmax/argmin. Grid (64,4): block = 64-row strip x 1024-column quarter.
// score s_ij = ww[j] - 2*g_ij (monotone in w_dis[i,j] for fixed i).
// Per-lane running packed bests in registers; one cross-lane reduction and
// global u64 atomic merge per wave at the end.
// ---------------------------------------------------------------------------
__global__ __launch_bounds__(256)
void gram_argextrema_kernel(const __bf16* __restrict__ wHi,
                            const __bf16* __restrict__ wLo,
                            const int* __restrict__ targets,
                            const float* __restrict__ ww,
                            u64* __restrict__ pmax,
                            u64* __restrict__ pmin) {
  __shared__ int   s_t[QCOLS];
  __shared__ float s_ww[QCOLS];

  const int tid  = threadIdx.x;
  const int lane = tid & 31;
  const int wave = tid >> 5;
  const int rowBase = blockIdx.x * 64;
  const int qBase   = blockIdx.y * QCOLS;

  for (int i = tid; i < QCOLS; i += 256) {
    s_t[i]  = targets[qBase + i];
    s_ww[i] = ww[qBase + i];
  }

  const int mi      = wave & 3;        // 4 row sub-tiles of 16
  const int nig     = wave >> 2;       // ni pair {0,1} or {2,3}
  const int kb      = (lane >> 4) * 8; // fragment K phase per half-wave
  const int halfOff = (lane >> 4) * 8; // C layout: lanes>=16 hold M+8

  const int mRow = rowBase + mi * 16 + (lane & 15);
  const __bf16* aRowH = wHi + (size_t)mRow * DW;
  const __bf16* aRowL = wLo + (size_t)mRow * DW;

  // Row targets for this lane's 8 C rows (constant across the sweep).
  int trow[8];
#pragma unroll
  for (int r = 0; r < 8; ++r) trow[r] = targets[rowBase + mi * 16 + halfOff + r];

  // Per-lane running packed (ordered-value, index) bests for 8 C rows.
  u64 bmax[8], bmin[8];
#pragma unroll
  for (int r = 0; r < 8; ++r) { bmax[r] = 0ull; bmin[r] = ~0ull; }

  __syncthreads();

  for (int ct = 0; ct < QCOLS / 64; ++ct) {
    const int n0 = qBase + ct * 64;
    const int nCol0 = n0 + (nig * 2 + 0) * 16 + (lane & 15);
    const int nCol1 = n0 + (nig * 2 + 1) * 16 + (lane & 15);
    const __bf16* b0H = wHi + (size_t)nCol0 * DW;
    const __bf16* b0L = wLo + (size_t)nCol0 * DW;
    const __bf16* b1H = wHi + (size_t)nCol1 * DW;
    const __bf16* b1L = wLo + (size_t)nCol1 * DW;

    if (ct < QCOLS / 64 - 1) {  // prefetch next tile's B rows (global_prefetch_b8)
      __builtin_prefetch(b0H + (size_t)64 * DW, 0, 1);
      __builtin_prefetch(b1H + (size_t)64 * DW, 0, 1);
      __builtin_prefetch(b0L + (size_t)64 * DW, 0, 1);
      __builtin_prefetch(b1L + (size_t)64 * DW, 0, 1);
    }

    v8f acc0 = {};
    v8f acc1 = {};
    for (int k0 = 0; k0 < DW; k0 += 32) {
      v16bf aH = load_frag_bf(aRowH, k0 + kb);
      v16bf aL = load_frag_bf(aRowL, k0 + kb);
      v16bf bH = load_frag_bf(b0H, k0 + kb);
      v16bf bL = load_frag_bf(b0L, k0 + kb);
      v16bf cH = load_frag_bf(b1H, k0 + kb);
      v16bf cL = load_frag_bf(b1L, k0 + kb);
      // Alternate accumulators: dependency distance 2 between WMMAs on each acc.
      acc0 = __builtin_amdgcn_wmma_f32_16x16x32_bf16(false, aH, false, bH,
                                                     (short)0, acc0, false, false);
      acc1 = __builtin_amdgcn_wmma_f32_16x16x32_bf16(false, aH, false, cH,
                                                     (short)0, acc1, false, false);
      acc0 = __builtin_amdgcn_wmma_f32_16x16x32_bf16(false, aH, false, bL,
                                                     (short)0, acc0, false, false);
      acc1 = __builtin_amdgcn_wmma_f32_16x16x32_bf16(false, aH, false, cL,
                                                     (short)0, acc1, false, false);
      acc0 = __builtin_amdgcn_wmma_f32_16x16x32_bf16(false, aL, false, bH,
                                                     (short)0, acc0, false, false);
      acc1 = __builtin_amdgcn_wmma_f32_16x16x32_bf16(false, aL, false, cH,
                                                     (short)0, acc1, false, false);
    }

    // Fold both 16x16 sub-tiles into the running per-row bests (registers only).
    const float wwj0 = s_ww[nCol0 - qBase];
    const int   tj0  = s_t[nCol0 - qBase];
    const float wwj1 = s_ww[nCol1 - qBase];
    const int   tj1  = s_t[nCol1 - qBase];
#pragma unroll
    for (int r = 0; r < 8; ++r) {
      {
        const bool ok = (trow[r] == tj0);
        const float s = __builtin_fmaf(-2.0f, acc0[r], wwj0);
        const unsigned u = ord_u32(s);
        u64 kmax = ok ? (((u64)u << 32) |
                         (unsigned)(0xFFFFFFFFu - (unsigned)nCol0)) : 0ull;
        u64 kmin = ok ? (((u64)u << 32) | (unsigned)nCol0) : ~0ull;
        bmax[r] = kmax > bmax[r] ? kmax : bmax[r];
        bmin[r] = kmin < bmin[r] ? kmin : bmin[r];
      }
      {
        const bool ok = (trow[r] == tj1);
        const float s = __builtin_fmaf(-2.0f, acc1[r], wwj1);
        const unsigned u = ord_u32(s);
        u64 kmax = ok ? (((u64)u << 32) |
                         (unsigned)(0xFFFFFFFFu - (unsigned)nCol1)) : 0ull;
        u64 kmin = ok ? (((u64)u << 32) | (unsigned)nCol1) : ~0ull;
        bmax[r] = kmax > bmax[r] ? kmax : bmax[r];
        bmin[r] = kmin < bmin[r] ? kmin : bmin[r];
      }
    }
  }

  // One cross-lane reduction (16 N-lanes) per wave, then merge partial bests
  // across the 4 column-quarter blocks via global u64 atomics.
#pragma unroll
  for (int r = 0; r < 8; ++r) {
    u64 kmax = bmax[r];
    u64 kmin = bmin[r];
#pragma unroll
    for (int m = 1; m <= 8; m <<= 1) {
      u64 pmaxv = shfl_xor_u64(kmax, m);
      u64 pminv = shfl_xor_u64(kmin, m);
      kmax = (pmaxv > kmax) ? pmaxv : kmax;
      kmin = (pminv < kmin) ? pminv : kmin;
    }
    if ((lane & 15) == 0) {
      const int grow = rowBase + mi * 16 + halfOff + r;
      atomicMax(&pmax[grow], kmax);   // global_atomic_max_u64
      atomicMin(&pmin[grow], kmin);   // global_atomic_min_u64
    }
  }
}

// ---------------------------------------------------------------------------
// Kernel 3: decode packed bests, exact fp32 distances for the two selected
// pairs per row + margin. One wave per row.
// ---------------------------------------------------------------------------
__global__ __launch_bounds__(256)
void pair_loss_kernel(const float* __restrict__ x,
                      const float* __restrict__ xx,
                      const u64* __restrict__ pmax,
                      const u64* __restrict__ pmin,
                      float* __restrict__ rowloss) {
  const int lane = threadIdx.x & 31;
  const int wave = threadIdx.x >> 5;
  const int row  = blockIdx.x * 8 + wave;
  const int ja = (int)(0xFFFFFFFFu - (unsigned)(pmax[row] & 0xFFFFFFFFull));
  const int jn = (int)(pmin[row] & 0xFFFFFFFFull);
  const float* xi = x + (size_t)row * DD;
  const float* xa = x + (size_t)ja  * DD;
  const float* xn = x + (size_t)jn  * DD;
  float da = 0.f, dn = 0.f;
  for (int k = lane * 4; k < DD; k += 128) {
    float4 v = *(const float4*)(xi + k);
    float4 a = *(const float4*)(xa + k);
    float4 b = *(const float4*)(xn + k);
    da += v.x * a.x + v.y * a.y + v.z * a.z + v.w * a.w;
    dn += v.x * b.x + v.y * b.y + v.z * b.z + v.w * b.w;
  }
#pragma unroll
  for (int m = 16; m >= 1; m >>= 1) {
    da += __shfl_xor(da, m, 32);
    dn += __shfl_xor(dn, m, 32);
  }
  if (lane == 0) {
    float dap2 = xx[row] + xx[ja] - 2.f * da;
    float dan2 = xx[row] + xx[jn] - 2.f * dn;
    dap2 = dap2 > 1e-12f ? dap2 : 1e-12f;
    dan2 = dan2 > 1e-12f ? dan2 : 1e-12f;
    float l = sqrtf(dap2) - sqrtf(dan2) + MARGIN_F;
    rowloss[row] = l > 0.f ? l : 0.f;
  }
}

// ---------------------------------------------------------------------------
// Kernel 4: mean over 4096 row losses -> scalar output.
// ---------------------------------------------------------------------------
__global__ __launch_bounds__(256)
void reduce_mean_kernel(const float* __restrict__ rowloss, float* __restrict__ out) {
  const int tid = threadIdx.x;
  float s = 0.f;
  for (int i = tid; i < NN; i += 256) s += rowloss[i];
#pragma unroll
  for (int m = 16; m >= 1; m >>= 1) s += __shfl_xor(s, m, 32);
  __shared__ float partial[8];
  if ((tid & 31) == 0) partial[tid >> 5] = s;
  __syncthreads();
  if (tid == 0) {
    float t = 0.f;
#pragma unroll
    for (int i = 0; i < 8; ++i) t += partial[i];
    out[0] = t / (float)NN;
  }
}

extern "C" void kernel_launch(void* const* d_in, const int* in_sizes, int n_in,
                              void* d_out, int out_size, void* d_ws, size_t ws_size,
                              hipStream_t stream) {
  (void)in_sizes; (void)n_in; (void)out_size; (void)ws_size;
  const float* inputs  = (const float*)d_in[0];
  const int*   targets = (const int*)d_in[1];
  const float* w       = (const float*)d_in[2];   // weights_vector[0]

  // Workspace layout (~8.2 MB):
  float*  xx      = (float*)d_ws;                     // [4096]
  float*  ww      = xx + NN;                          // [4096]
  float*  rowloss = ww + NN;                          // [4096]
  u64*    pmax    = (u64*)(rowloss + NN);             // [4096] packed argmax
  u64*    pmin    = pmax + NN;                        // [4096] packed argmin
  __bf16* w_hi    = (__bf16*)(pmin + NN);             // [4096*512] (4 MB)
  __bf16* w_lo    = w_hi + (size_t)NN * DW;           // [4096*512] (4 MB)

  init_best_kernel<<<NN / 256, 256, 0, stream>>>(pmax, pmin);
  norms_split_kernel<<<NN, 256, 0, stream>>>(inputs, w, xx, ww, w_hi, w_lo);
  dim3 grid2(NN / 64, NN / QCOLS);   // 64 row strips x 4 column quarters
  gram_argextrema_kernel<<<grid2, 256, 0, stream>>>(w_hi, w_lo, targets, ww,
                                                    pmax, pmin);
  pair_loss_kernel<<<NN / 8, 256, 0, stream>>>(inputs, xx, pmax, pmin, rowloss);
  reduce_mean_kernel<<<1, 256, 0, stream>>>(rowloss, (float*)d_out);
}
